// ExternalMemory_78108275245215
// MI455X (gfx1250) — matrix operand order repaired and checked by reading, not compiled, not run on hardware
//
#include <hip/hip_runtime.h>

typedef __attribute__((ext_vector_type(16))) __bf16 v16bf;
typedef __attribute__((ext_vector_type(8)))  float  v8f;

#define B_DIM 65536
#define H_DIM 1024
#define S_DIM 1024
#define D_DIM 128
#define MT    128     // M rows per workgroup
#define NC    128     // N (slot) chunk
#define KT    64      // K stage
#define LDSS  72      // padded LDS stride in bf16 elems (144B = 36 banks)
#define NTHR  512     // 16 waves: 8 row slabs x 2 column halves

#define OVL_OFF    ((size_t)MT * S_DIM * 2)            // 256KB write-logits cache
#define STATS_OFF  (OVL_OFF + 3u * MT * LDSS * 2)      // +54KB staging overlap
#define STATS_FLTS (4 * MT + 4 * 2 * MT + 2 * MT)      // running + partials + published
#define SMEM_BYTES (STATS_OFF + STATS_FLTS * 4)        // = 324608 (<= 320KB)

// sched_group_barrier masks
#define SG_MFMA    0x008   // matches v_wmma on gfx12 (isMFMAorWMMA)
#define SG_DSREAD  0x100

// ---------------- fragment loaders (ISA 7.12.2 layouts) ----------------
// A (16x32 bf16): lanes 0-15 -> rows 0-15, K {0..7, 16..23}; lanes 16-31 same rows, K {8..15, 24..31}
static __device__ __forceinline__ v16bf load_frag_a(const __bf16* st, int row_base, int ko,
                                                    int idx, int hl) {
  union { v16bf v; uint4 q[2]; } f;
  const __bf16* base = st + (row_base + idx) * LDSS + ko + hl * 8;
  f.q[0] = *(const uint4*)(base);
  f.q[1] = *(const uint4*)(base + 16);
  return f.v;
}
// B (32x16 bf16): lanes 0-15 -> cols 0-15, K {0..7, 16..23}; lanes 16-31 same cols, K {8..15, 24..31}
static __device__ __forceinline__ v16bf load_frag_b(const __bf16* st, int col_base, int ko,
                                                    int idx, int hl) {
  union { v16bf v; uint4 q[2]; } f;
  const __bf16* base = st + (col_base + idx) * LDSS + ko + hl * 8;
  f.q[0] = *(const uint4*)(base);
  f.q[1] = *(const uint4*)(base + 16);
  return f.v;
}

// reductions across the 16-lane half that owns one matrix row
static __device__ __forceinline__ float half_max(float x) {
#pragma unroll
  for (int m = 1; m < 16; m <<= 1) x = fmaxf(x, __shfl_xor(x, m, 32));
  return x;
}
static __device__ __forceinline__ float half_sum(float x) {
#pragma unroll
  for (int m = 1; m < 16; m <<= 1) x += __shfl_xor(x, m, 32);
  return x;
}

// =======================================================================
// Fused: read/write logits GEMMs (WMMA bf16), flash-softmax read path
// (O = softmax(Q@Wr+b) @ memory), write logits cached in LDS then reduced
// into w_sum[S]. 16 waves: each wave = 16 rows x 64 logit columns
// (and 16 rows x 64 output-D columns in the P@memory phase).
// =======================================================================
__global__ __launch_bounds__(NTHR, 1)
void fused_attn_kernel(const float* __restrict__ query,
                       const float* __restrict__ memory,
                       const float* __restrict__ W_read,  const float* __restrict__ b_read,
                       const float* __restrict__ W_write, const float* __restrict__ b_write,
                       float* __restrict__ read_out, float* __restrict__ w_sum) {
  extern __shared__ char smem[];
  __bf16* lw  = (__bf16*)smem;                         // [MT][S_DIM] write logits
  char*   ovl = smem + OVL_OFF;                        // overlapped staging region
  __bf16* Ast = (__bf16*)ovl;                          // [MT][LDSS]   phase A
  __bf16* Brs = (__bf16*)(ovl + 1 * MT * LDSS * 2);    // [NC][LDSS]
  __bf16* Bws = (__bf16*)(ovl + 2 * MT * LDSS * 2);    // [NC][LDSS]
  __bf16* Pst = (__bf16*)ovl;                          // [MT][LDSS]   phase B (alias)
  __bf16* Mst = (__bf16*)(ovl + 1 * MT * LDSS * 2);    // [D_DIM][LDSS]
  float*  s_mr  = (float*)(smem + STATS_OFF);          // [MT] running max (read)
  float*  s_zr  = s_mr  + MT;                          // [MT] running sum (read)
  float*  s_mw  = s_zr  + MT;                          // [MT] running max (write)
  float*  s_zw  = s_mw  + MT;                          // [MT] running sum (write)
  float*  s_pxr = s_zw  + MT;                          // [2][MT] partial max (read)
  float*  s_psr = s_pxr + 2 * MT;                      // [2][MT] partial sum (read)
  float*  s_pxw = s_psr + 2 * MT;                      // [2][MT] partial max (write)
  float*  s_psw = s_pxw + 2 * MT;                      // [2][MT] partial sum (write)
  float*  s_mn  = s_psw + 2 * MT;                      // [MT] published m_new (read)
  float*  s_co  = s_mn  + MT;                          // [MT] published corr (read)

  const int tid  = threadIdx.x;
  const int wave = tid >> 5;
  const int rw   = wave & 7;    // row slab
  const int ch   = wave >> 3;   // column half (0 or 1)
  const int lane = tid & 31;
  const int idx  = lane & 15;   // N (or M for A-frags) within 16-tile
  const int hl   = lane >> 4;   // lane half
  const int rowg0 = blockIdx.x * MT;
  const int wrow  = rw * 16;

  if (tid < MT) { s_mr[tid] = -3.0e38f; s_zr[tid] = 0.f; s_mw[tid] = -3.0e38f; s_zw[tid] = 0.f; }
  // (ordered before first use by the staging barriers below)

  v8f Or[4];
#pragma unroll
  for (int t = 0; t < 4; ++t)
#pragma unroll
    for (int v = 0; v < 8; ++v) Or[t][v] = 0.f;

  for (int j = 0; j < S_DIM / NC; ++j) {
    const int n0 = j * NC;
    v8f Cr[4], Cw[4];
#pragma unroll
    for (int t = 0; t < 4; ++t)
#pragma unroll
      for (int v = 0; v < 8; ++v) { Cr[t][v] = 0.f; Cw[t][v] = 0.f; }

    // ---------------- phase A: logits GEMMs over K = H ----------------
    for (int kk = 0; kk < H_DIM / KT; ++kk) {
      const int k0 = kk * KT;
      __syncthreads();
      // stage A tile (fp32 -> bf16), float4 loads, k fastest => coalesced
      for (int e = tid; e < MT * KT / 4; e += NTHR) {
        int r = e >> 4, k4 = (e & 15) * 4;
        float4 qv = *(const float4*)(query + (size_t)(rowg0 + r) * H_DIM + k0 + k4);
        __bf16* dst = Ast + r * LDSS + k4;
        dst[0] = (__bf16)qv.x; dst[1] = (__bf16)qv.y;
        dst[2] = (__bf16)qv.z; dst[3] = (__bf16)qv.w;
      }
      // stage Wr/Ww tiles transposed to [n][k]
      for (int e = tid; e < KT * NC; e += NTHR) {
        int k = e >> 7, n = e & 127;
        Brs[n * LDSS + k] = (__bf16)W_read [(size_t)(k0 + k) * S_DIM + n0 + n];
        Bws[n * LDSS + k] = (__bf16)W_write[(size_t)(k0 + k) * S_DIM + n0 + n];
      }
      __syncthreads();
      v16bf a0 = load_frag_a(Ast, wrow, 0,  idx, hl);
      v16bf a1 = load_frag_a(Ast, wrow, 32, idx, hl);
      // software-pipelined over the 4 column subtiles; the sched_group_barrier
      // sequence below pins the scheduler to this shape pre-RA so the WAR
      // chains are not re-serialized into a single fragment buffer.
      v16bf br0c, bw0c, br1c, bw1c, br0n, bw0n, br1n, bw1n;
      {
        const int cb = (4 * ch) * 16;
        br0c = load_frag_b(Brs, cb, 0,  idx, hl);
        bw0c = load_frag_b(Bws, cb, 0,  idx, hl);
        br1c = load_frag_b(Brs, cb, 32, idx, hl);
        bw1c = load_frag_b(Bws, cb, 32, idx, hl);
      }
#pragma unroll
      for (int t = 0; t < 4; ++t) {
        if (t < 3) {
          const int nb = (4 * ch + t + 1) * 16;
          br0n = load_frag_b(Brs, nb, 0,  idx, hl);
          bw0n = load_frag_b(Bws, nb, 0,  idx, hl);
          br1n = load_frag_b(Brs, nb, 32, idx, hl);
          bw1n = load_frag_b(Bws, nb, 32, idx, hl);
        }
        Cr[t] = __builtin_amdgcn_wmma_f32_16x16x32_bf16(false, a0, false, br0c, (short)0, Cr[t], false, false);
        Cw[t] = __builtin_amdgcn_wmma_f32_16x16x32_bf16(false, a0, false, bw0c, (short)0, Cw[t], false, false);
        Cr[t] = __builtin_amdgcn_wmma_f32_16x16x32_bf16(false, a1, false, br1c, (short)0, Cr[t], false, false);
        Cw[t] = __builtin_amdgcn_wmma_f32_16x16x32_bf16(false, a1, false, bw1c, (short)0, Cw[t], false, false);
        br0c = br0n; bw0c = bw0n; br1c = br1n; bw1c = bw1n;
      }
      // Enforce: [A + t0 + t1 loads] -> t0 WMMAs -> [t2 loads] -> t1 WMMAs
      //          -> [t3 loads] -> t2+t3 WMMAs   (36 DS reads, 16 WMMAs total)
      __builtin_amdgcn_sched_group_barrier(SG_DSREAD, 20, 0);
      __builtin_amdgcn_sched_group_barrier(SG_MFMA,    4, 0);
      __builtin_amdgcn_sched_group_barrier(SG_DSREAD,  8, 0);
      __builtin_amdgcn_sched_group_barrier(SG_MFMA,    4, 0);
      __builtin_amdgcn_sched_group_barrier(SG_DSREAD,  8, 0);
      __builtin_amdgcn_sched_group_barrier(SG_MFMA,    8, 0);
    }

    // ---------------- bias add ----------------
#pragma unroll
    for (int t = 0; t < 4; ++t) {
      float br = b_read [n0 + (4 * ch + t) * 16 + idx];
      float bw = b_write[n0 + (4 * ch + t) * 16 + idx];
#pragma unroll
      for (int v = 0; v < 8; ++v) { Cr[t][v] += br; Cw[t][v] += bw; }
    }

    // ---------------- per-wave partial softmax stats -> LDS ----------------
#pragma unroll
    for (int v = 0; v < 8; ++v) {
      int r = wrow + v + 8 * hl;
      float pmw = Cw[0][v], pmr = Cr[0][v];
#pragma unroll
      for (int t = 1; t < 4; ++t) { pmw = fmaxf(pmw, Cw[t][v]); pmr = fmaxf(pmr, Cr[t][v]); }
      pmw = half_max(pmw); pmr = half_max(pmr);
      float psw = 0.f, psr = 0.f;
#pragma unroll
      for (int t = 0; t < 4; ++t) { psw += __expf(Cw[t][v] - pmw); psr += __expf(Cr[t][v] - pmr); }
      psw = half_sum(psw); psr = half_sum(psr);
      if (idx == 0) {
        s_pxw[ch * MT + r] = pmw; s_psw[ch * MT + r] = psw;
        s_pxr[ch * MT + r] = pmr; s_psr[ch * MT + r] = psr;
      }
    }
    // cache raw write logits (independent of the combine)
#pragma unroll
    for (int t = 0; t < 4; ++t)
#pragma unroll
      for (int v = 0; v < 8; ++v)
        lw[(size_t)(wrow + v + 8 * hl) * S_DIM + n0 + (4 * ch + t) * 16 + idx] = (__bf16)Cw[t][v];
    __syncthreads();

    // ---------------- owner waves combine + update running stats ----------------
    if (ch == 0) {
#pragma unroll
      for (int v = 0; v < 8; ++v) {
        int r = wrow + v + 8 * hl;
        // write path
        float pm0 = s_pxw[r], pm1 = s_pxw[MT + r];
        float cmax = fmaxf(pm0, pm1);
        float csum = s_psw[r] * __expf(pm0 - cmax) + s_psw[MT + r] * __expf(pm1 - cmax);
        float mold = s_mw[r];
        float mn   = fmaxf(mold, cmax);
        float znew = s_zw[r] * __expf(mold - mn) + csum * __expf(cmax - mn);
        // read path
        float qm0 = s_pxr[r], qm1 = s_pxr[MT + r];
        float rmax = fmaxf(qm0, qm1);
        float rsum = s_psr[r] * __expf(qm0 - rmax) + s_psr[MT + r] * __expf(qm1 - rmax);
        float rold = s_mr[r];
        float rmn  = fmaxf(rold, rmax);
        float corr = __expf(rold - rmn);
        float rz   = s_zr[r] * corr + rsum * __expf(rmax - rmn);
        if (idx == 0) {
          s_mw[r] = mn;  s_zw[r] = znew;
          s_mr[r] = rmn; s_zr[r] = rz;
          s_mn[r] = rmn; s_co[r] = corr;
        }
      }
    }
    __syncthreads();

    // ---------------- all waves: rescale O, Cr -> P ----------------
#pragma unroll
    for (int v = 0; v < 8; ++v) {
      int r = wrow + v + 8 * hl;
      float mnr  = s_mn[r];
      float corr = s_co[r];
#pragma unroll
      for (int t = 0; t < 4; ++t) { Cr[t][v] = __expf(Cr[t][v] - mnr); Or[t][v] *= corr; }
    }

    // ---------------- O += P @ memory, K = 128 in two 64-halves ----------------
#pragma unroll
    for (int h = 0; h < 2; ++h) {
      __syncthreads();   // phase-A/-B LDS regions alias: everyone done reading
      if (ch == h) {     // this wave owns the slots of half h
#pragma unroll
        for (int t = 0; t < 4; ++t)
#pragma unroll
          for (int v = 0; v < 8; ++v)
            Pst[(wrow + v + 8 * hl) * LDSS + t * 16 + idx] = (__bf16)Cr[t][v];
      }
      for (int e = tid; e < 64 * D_DIM; e += NTHR) {      // memory chunk -> [d][slot]
        int sl = e >> 7, d = e & 127;
        Mst[d * LDSS + sl] = (__bf16)memory[(size_t)(n0 + 64 * h + sl) * D_DIM + d];
      }
      __syncthreads();
      v16bf pa0 = load_frag_a(Pst, wrow, 0,  idx, hl);
      v16bf pa1 = load_frag_a(Pst, wrow, 32, idx, hl);
#pragma unroll
      for (int t = 0; t < 4; ++t) {
        const int db = (4 * ch + t) * 16;
        v16bf mb0 = load_frag_b(Mst, db, 0,  idx, hl);
        v16bf mb1 = load_frag_b(Mst, db, 32, idx, hl);
        Or[t] = __builtin_amdgcn_wmma_f32_16x16x32_bf16(false, pa0, false, mb0, (short)0, Or[t], false, false);
        Or[t] = __builtin_amdgcn_wmma_f32_16x16x32_bf16(false, pa1, false, mb1, (short)0, Or[t], false, false);
      }
    }
  }

  // ---------------- emit read_content = O / Z (disjoint D columns) ----------------
#pragma unroll
  for (int v = 0; v < 8; ++v) {
    int r = wrow + v + 8 * hl;
    float zi = 1.0f / s_zr[r];
    size_t grow = (size_t)(rowg0 + r);
#pragma unroll
    for (int t = 0; t < 4; ++t)
      read_out[grow * D_DIM + (4 * ch + t) * 16 + idx] = Or[t][v] * zi;
  }

  // ---------------- write path: sum_b exp(l - m)/Z into w_sum ----------------
  __syncthreads();
  if (tid < MT) s_zw[tid] = 1.0f / s_zw[tid];
  __syncthreads();
  for (int i = 0; i < S_DIM / NTHR; ++i) {
    int s = i * NTHR + tid;
    float acc = 0.f;
    for (int r = 0; r < MT; ++r) {
      float l = (float)lw[(size_t)r * S_DIM + s];
      acc += __expf(l - s_mw[r]) * s_zw[r];
    }
    __hip_atomic_fetch_add(&w_sum[s], acc, __ATOMIC_RELAXED, __HIP_MEMORY_SCOPE_AGENT);
  }
}

// =======================================================================
// content column sums: h_sum[h] = sum_b content[b][h]
// =======================================================================
__global__ __launch_bounds__(256)
void colsum_kernel(const float* __restrict__ content, float* __restrict__ h_sum) {
  int h  = blockIdx.x * 256 + threadIdx.x;   // gridDim.x = H/256
  int b0 = blockIdx.y * 256;                 // gridDim.y = B/256
  float acc = 0.f;
  for (int b = 0; b < 256; ++b) acc += content[(size_t)(b0 + b) * H_DIM + h];
  __hip_atomic_fetch_add(&h_sum[h], acc, __ATOMIC_RELAXED, __HIP_MEMORY_SCOPE_AGENT);
}

// c_mean[d] = (h_sum/B) @ W_content + b_content
__global__ __launch_bounds__(128)
void cmean_kernel(const float* __restrict__ h_sum, const float* __restrict__ W_content,
                  const float* __restrict__ b_content, float* __restrict__ c_mean) {
  int d = threadIdx.x;
  float acc = 0.f;
  for (int hh = 0; hh < H_DIM; ++hh) acc += h_sum[hh] * W_content[(size_t)hh * D_DIM + d];
  c_mean[d] = acc * (1.0f / (float)B_DIM) + b_content[d];
}

// memory / age update
__global__ __launch_bounds__(256)
void finalize_kernel(const float* __restrict__ memory, const float* __restrict__ memory_age,
                     const float* __restrict__ w_sum, const float* __restrict__ c_mean,
                     float* __restrict__ out_mem, float* __restrict__ out_age) {
  int i = blockIdx.x * 256 + threadIdx.x;    // S*D threads
  int s = i >> 7, d = i & 127;
  float wm   = w_sum[s] * (1.0f / (float)B_DIM);
  bool  mask = wm > 0.01f;
  float cons = 1.0f / (1.0f + __expf(-0.1f * memory_age[s]));
  float f    = wm * cons;
  float m    = memory[i];
  out_mem[i] = mask ? (1.f - f) * m + f * c_mean[d] : m;
  if (d == 0) out_age[s] = memory_age[s] + (mask ? 1.f : 0.f);
}

extern "C" void kernel_launch(void* const* d_in, const int* in_sizes, int n_in,
                              void* d_out, int out_size, void* d_ws, size_t ws_size,
                              hipStream_t stream) {
  const float* query      = (const float*)d_in[0];
  const float* content    = (const float*)d_in[1];
  const float* memory     = (const float*)d_in[2];
  const float* memory_age = (const float*)d_in[3];
  const float* W_read     = (const float*)d_in[4];
  const float* b_read     = (const float*)d_in[5];
  const float* W_write    = (const float*)d_in[6];
  const float* b_write    = (const float*)d_in[7];
  const float* W_content  = (const float*)d_in[8];
  const float* b_content  = (const float*)d_in[9];

  float* out      = (float*)d_out;
  float* read_out = out;                                   // [B, D]
  float* out_mem  = out + (size_t)B_DIM * D_DIM;           // [S, D]
  float* out_age  = out_mem + (size_t)S_DIM * D_DIM;       // [S]

  float* w_sum  = (float*)d_ws;          // [S]
  float* h_sum  = w_sum + S_DIM;         // [H]
  float* c_mean = h_sum + H_DIM;         // [D]

  hipMemsetAsync(d_ws, 0, (size_t)(S_DIM + H_DIM) * sizeof(float), stream);

  fused_attn_kernel<<<B_DIM / MT, NTHR, SMEM_BYTES, stream>>>(query, memory, W_read, b_read,
                                                              W_write, b_write, read_out, w_sum);
  colsum_kernel<<<dim3(H_DIM / 256, B_DIM / 256), 256, 0, stream>>>(content, h_sum);
  cmean_kernel<<<1, 128, 0, stream>>>(h_sum, W_content, b_content, c_mean);
  finalize_kernel<<<(S_DIM * D_DIM) / 256, 256, 0, stream>>>(memory, memory_age, w_sum,
                                                             c_mean, out_mem, out_age);
}
